// NetIzhi_37340445671935
// MI455X (gfx1250) — matrix operand order repaired and checked by reading, not compiled, not run on hardware
//
#include <hip/hip_runtime.h>

// ---------------------------------------------------------------------------
// Izhikevich SNN for MI455X (gfx1250, wave32, WMMA + async global->LDS).
//
//   K1-3: split W1, x, W2 into f16 hi/lo pairs (hi = f16(w), lo = f16(w-hi))
//   K4  : GEMM1  cur1[256,4096] = x @ W1^T + b1   (split-f16 WMMA, 3 terms)
//   K5  : layer-1 dynamics (no feedback), all 100 steps -> SPK1[25600,4096] f16
//   K6  : GEMM2  CUR2[25600,512] = SPK1 @ W2^T + b2
//          - B (W2 hi/lo) staged in LDS via global_load_async_to_lds_b128,
//            double-buffered, shared by 8 waves (8x L2 traffic cut)
//   K7  : layer-2 scan per (b,o) lane -> d_out = [spk2_rec | mem2_rec]
// ---------------------------------------------------------------------------

typedef __attribute__((ext_vector_type(16))) _Float16 v16h;
typedef __attribute__((ext_vector_type(8)))  _Float16 v8h;
typedef __attribute__((ext_vector_type(8)))  float    v8f;

union AV { v16h v; v8h h[2]; };

__device__ __forceinline__ v8f wmma16(v16h a, v16h b, v8f c) {
  return __builtin_amdgcn_wmma_f32_16x16x32_f16(
      false, a, false, b, (short)0, c, false, false);
}

// Low 32 bits of a flat pointer into the LDS aperture == LDS byte offset.
__device__ __forceinline__ unsigned lds_off(const void* p) {
  return (unsigned)(unsigned long long)p;
}

// gfx1250 async DMA: 16 bytes per lane, memory -> LDS, tracked by ASYNCcnt.
__device__ __forceinline__ void async_ld16(unsigned lds_dst, const void* gsrc) {
  asm volatile("global_load_async_to_lds_b128 %0, %1, off"
               :: "v"(lds_dst), "v"(gsrc) : "memory");
}

// ------------------------- K1-3: f32 -> f16 hi/lo split ---------------------
__global__ void split_f16(const float* __restrict__ src,
                          _Float16* __restrict__ hi,
                          _Float16* __restrict__ lo, int n) {
  int i = blockIdx.x * blockDim.x + threadIdx.x;
  if (i < n) {
    float v = src[i];
    _Float16 h = (_Float16)v;
    hi[i] = h;
    lo[i] = (_Float16)(v - (float)h);
  }
}

// ------------------------- K4: cur1 = x @ W1^T + b1 -------------------------
__global__ void gemm1_kernel(const _Float16* __restrict__ xh,
                             const _Float16* __restrict__ xl,
                             const _Float16* __restrict__ w1h,
                             const _Float16* __restrict__ w1l,
                             const float* __restrict__ b1,
                             float* __restrict__ cur1) {
  const int NI = 2048, NH = 4096;
  const int warp = threadIdx.x >> 5;
  const int lane = threadIdx.x & 31;
  const int r = lane & 15;
  const int hb = lane >> 4;
  const int wid = blockIdx.x * 8 + warp; // 0..1023
  const int mt = wid >> 6;               // 0..15
  const int nb = (wid & 63) * 64;

  v8f acc[4];
#pragma unroll
  for (int j = 0; j < 4; ++j) {
    float bias = b1[nb + j * 16 + r];
#pragma unroll
    for (int v = 0; v < 8; ++v) acc[j][v] = bias;
  }

  const _Float16* xhp = xh + (size_t)(mt * 16 + r) * NI;
  const _Float16* xlp = xl + (size_t)(mt * 16 + r) * NI;

  for (int kb = 0; kb < NI; kb += 32) {
    AV ah, al;
    ah.h[0] = *(const v8h*)(xhp + kb + hb * 16);
    ah.h[1] = *(const v8h*)(xhp + kb + hb * 16 + 8);
    al.h[0] = *(const v8h*)(xlp + kb + hb * 16);
    al.h[1] = *(const v8h*)(xlp + kb + hb * 16 + 8);
#pragma unroll
    for (int j = 0; j < 4; ++j) {
      size_t boff = (size_t)(nb + j * 16 + r) * NI + kb + hb * 16;
      v16h bh = *(const v16h*)(w1h + boff);
      v16h bl = *(const v16h*)(w1l + boff);
      acc[j] = wmma16(ah.v, bh, acc[j]);
      acc[j] = wmma16(ah.v, bl, acc[j]);
      acc[j] = wmma16(al.v, bh, acc[j]);
    }
  }

#pragma unroll
  for (int j = 0; j < 4; ++j)
#pragma unroll
    for (int v = 0; v < 8; ++v) {
      int m = mt * 16 + v + hb * 8;
      cur1[(size_t)m * NH + nb + j * 16 + r] = acc[j][v];
    }
}

// -------------- K5: layer-1 Izhikevich, all 100 steps, no feedback ----------
__global__ void layer1_kernel(const float* __restrict__ cur1,
                              _Float16* __restrict__ spk1) {
  const int idx = blockIdx.x * blockDim.x + threadIdx.x;
  const float cur = cur1[idx];
  float mem = 0.0f, rec = 0.0f;
  for (int t = 0; t < 100; ++t) {
    float spk = (mem >= 30.0f) ? 1.0f : 0.0f;
    spk1[(size_t)t * 1048576 + idx] = (_Float16)spk;
    float mem_n = mem + (0.04f * mem * mem + 5.0f * mem + 140.0f - rec + cur);
    float rec_n = rec + 0.02f * (0.2f * mem - rec);
    if (spk > 0.0f) { mem_n = -65.0f; rec_n += 8.0f; }
    mem = mem_n; rec = rec_n;
  }
}

// ------------- K6: CUR2[25600,512] = SPK1 @ W2^T + b2 (batched over T) ------
// grid = (8 n-strips, 100 m-blocks) x 256 threads (8 waves, 2 m-tiles each).
// B chunks (64n x 64k, hi+lo) double-buffered in LDS via async DMA.
#define KC      64              // k-chunk
#define BPITCH  72              // LDS row pitch in halves (64 + 8 pad)
__global__ void gemm2_kernel(const _Float16* __restrict__ spk1,
                             const _Float16* __restrict__ w2h,
                             const _Float16* __restrict__ w2l,
                             const float* __restrict__ b2,
                             float* __restrict__ cur2) {
  const int NH = 4096, NO = 512;
  __shared__ _Float16 sB[2][2][64 * BPITCH];   // [buf][hi/lo][n*pitch+k] 36 KB

  const int tid  = threadIdx.x;
  const int warp = tid >> 5;
  const int lane = tid & 31;
  const int r = lane & 15;
  const int hb = lane >> 4;
  const int nb = blockIdx.x * 64;
  const int mt0 = (blockIdx.y * 8 + warp) * 2;

  // cooperative fill mapping: 256 threads x (2 rows x 16B) per array
  const int frow = tid >> 3;          // 0..31 (and +32)
  const int fkg  = (tid & 7) * 8;     // halves, 16B granules
  const unsigned sh[2] = { lds_off(&sB[0][0][0]), lds_off(&sB[1][0][0]) };
  const unsigned sl[2] = { lds_off(&sB[0][1][0]), lds_off(&sB[1][1][0]) };

  v8f acc[2][4];
#pragma unroll
  for (int i = 0; i < 2; ++i)
#pragma unroll
    for (int j = 0; j < 4; ++j) {
      float bias = b2[nb + j * 16 + r];
#pragma unroll
      for (int v = 0; v < 8; ++v) acc[i][j][v] = bias;
    }

  const _Float16* a0p = spk1 + (size_t)(mt0 * 16 + r) * NH;
  const _Float16* a1p = spk1 + (size_t)((mt0 + 1) * 16 + r) * NH;

  // issue fill of chunk 0 into buffer 0
#pragma unroll
  for (int rr = 0; rr < 2; ++rr) {
    int n = frow + rr * 32;
    unsigned d = (unsigned)((n * BPITCH + fkg) * 2);
    async_ld16(sh[0] + d, (const void*)(w2h + (size_t)(nb + n) * NH + fkg));
    async_ld16(sl[0] + d, (const void*)(w2l + (size_t)(nb + n) * NH + fkg));
  }

  const int NCHUNK = NH / KC;  // 64
  for (int c = 0; c < NCHUNK; ++c) {
    const int kb = c * KC;
    if (c + 1 < NCHUNK) {
      const int buf = (c + 1) & 1;
      const int kn = kb + KC;
#pragma unroll
      for (int rr = 0; rr < 2; ++rr) {
        int n = frow + rr * 32;
        unsigned d = (unsigned)((n * BPITCH + fkg) * 2);
        async_ld16(sh[buf] + d, (const void*)(w2h + (size_t)(nb + n) * NH + kn + fkg));
        async_ld16(sl[buf] + d, (const void*)(w2l + (size_t)(nb + n) * NH + kn + fkg));
      }
      asm volatile("s_wait_asynccnt 0x4" ::: "memory");  // chunk c done, c+1 in flight
    } else {
      asm volatile("s_wait_asynccnt 0x0" ::: "memory");
    }
    __syncthreads();

    const _Float16* bh_s = &sB[c & 1][0][0];
    const _Float16* bl_s = &sB[c & 1][1][0];
#pragma unroll
    for (int ks = 0; ks < 2; ++ks) {       // two K=32 steps per chunk
      const int kc = ks * 32;
      AV a0, a1;
      a0.h[0] = *(const v8h*)(a0p + kb + kc + hb * 16);
      a0.h[1] = *(const v8h*)(a0p + kb + kc + hb * 16 + 8);
      a1.h[0] = *(const v8h*)(a1p + kb + kc + hb * 16);
      a1.h[1] = *(const v8h*)(a1p + kb + kc + hb * 16 + 8);
#pragma unroll
      for (int j = 0; j < 4; ++j) {
        int so = (j * 16 + r) * BPITCH + kc + hb * 16;
        AV bh, bl;
        bh.h[0] = *(const v8h*)(bh_s + so);
        bh.h[1] = *(const v8h*)(bh_s + so + 8);
        bl.h[0] = *(const v8h*)(bl_s + so);
        bl.h[1] = *(const v8h*)(bl_s + so + 8);
        acc[0][j] = wmma16(a0.v, bh.v, acc[0][j]);
        acc[0][j] = wmma16(a0.v, bl.v, acc[0][j]);
        acc[1][j] = wmma16(a1.v, bh.v, acc[1][j]);
        acc[1][j] = wmma16(a1.v, bl.v, acc[1][j]);
      }
    }
    __syncthreads();   // all waves done with buf c before it is refilled
  }

#pragma unroll
  for (int i = 0; i < 2; ++i)
#pragma unroll
    for (int j = 0; j < 4; ++j)
#pragma unroll
      for (int v = 0; v < 8; ++v) {
        int m = (mt0 + i) * 16 + v + hb * 8;
        cur2[(size_t)m * NO + nb + j * 16 + r] = acc[i][j][v];
      }
}

// ---------------- K7: layer-2 scan + output (131072 independent lanes) ------
__global__ void layer2_kernel(const float* __restrict__ cur2,
                              float* __restrict__ out) {
  const int idx = blockIdx.x * blockDim.x + threadIdx.x;
  const size_t MEMOFF = (size_t)100 * 131072;
  float mem = 0.0f, rec = 0.0f;
  for (int t = 0; t < 100; ++t) {
    float cur = cur2[(size_t)t * 131072 + idx];
    float spk = (mem >= 30.0f) ? 1.0f : 0.0f;
    float mem_n = mem + (0.04f * mem * mem + 5.0f * mem + 140.0f - rec + cur);
    float rec_n = rec + 0.02f * (0.2f * mem - rec);
    if (spk > 0.0f) { mem_n = -65.0f; rec_n += 8.0f; }
    out[(size_t)t * 131072 + idx] = spk;
    out[MEMOFF + (size_t)t * 131072 + idx] = mem_n;
    mem = mem_n; rec = rec_n;
  }
}

// ---------------------------------------------------------------------------
extern "C" void kernel_launch(void* const* d_in, const int* in_sizes, int n_in,
                              void* d_out, int out_size, void* d_ws, size_t ws_size,
                              hipStream_t stream) {
  const float* x  = (const float*)d_in[0];
  const float* W1 = (const float*)d_in[1];
  const float* b1 = (const float*)d_in[2];
  const float* W2 = (const float*)d_in[3];
  const float* b2 = (const float*)d_in[4];
  float* out = (float*)d_out;

  char* ws = (char*)d_ws;
  _Float16* W1h  = (_Float16*)(ws + 0);
  _Float16* W1l  = (_Float16*)(ws + 16777216);
  _Float16* xh   = (_Float16*)(ws + 33554432);
  _Float16* xl   = (_Float16*)(ws + 34603008);
  _Float16* W2h  = (_Float16*)(ws + 35651584);
  _Float16* W2l  = (_Float16*)(ws + 39845888);
  float*    cur1 = (float*)   (ws + 44040192);
  _Float16* SPK1 = (_Float16*)(ws + 48234496);
  float*    CUR2 = (float*)   (ws + 257949696);

  split_f16<<<32768, 256, 0, stream>>>(W1, W1h, W1l, 4096 * 2048);
  split_f16<<< 2048, 256, 0, stream>>>(x,  xh,  xl,  256 * 2048);
  split_f16<<< 8192, 256, 0, stream>>>(W2, W2h, W2l, 512 * 4096);

  gemm1_kernel<<<128, 256, 0, stream>>>(xh, xl, W1h, W1l, b1, cur1);
  layer1_kernel<<<4096, 256, 0, stream>>>(cur1, SPK1);
  gemm2_kernel<<<dim3(8, 100), 256, 0, stream>>>(SPK1, W2h, W2l, b2, CUR2);
  layer2_kernel<<<512, 256, 0, stream>>>(CUR2, out);
}
